// LinearCrossEntropy_15582141350149
// MI455X (gfx1250) — compile-verified, matrix operand order checked by reading, not run on hardware
//
#include <hip/hip_runtime.h>
#include <math.h>

#define Ntok 4096
#define Ddim 2048
#define Vvoc 65536
#define BM 128
#define BN 128
#define KC 64
#define KSTAGES (Ddim / KC)      // 32
#define SLICES 32
#define VSLICE (Vvoc / SLICES)   // 2048 columns per workgroup
#define LDA 72                   // bf16 elems per LDS row (64 + 8 pad -> 144B stride, conflict-free b128)
#define IGNORE_IDX (-100)

typedef __attribute__((ext_vector_type(16))) __bf16 v16bf;
typedef __attribute__((ext_vector_type(2)))  __bf16 v2bf;
typedef __attribute__((ext_vector_type(8)))  float  v8f;

union FragBF {
  v16bf v;
  uint4 u[2];
};

__device__ __forceinline__ unsigned short f2bf(float f) {
  unsigned int u = __float_as_uint(f);
  u += 0x7FFFu + ((u >> 16) & 1u);   // round-to-nearest-even
  return (unsigned short)(u >> 16);
}

// Pack two fp32 -> packed bf16x2 (RNE). Prefer the single-instruction
// v_cvt_pk_bf16_f32 when the builtin exists for this target.
__device__ __forceinline__ unsigned pack_bf16(float lo, float hi) {
#if __has_builtin(__builtin_amdgcn_cvt_pk_bf16_f32)
  union { v2bf v; unsigned u; } cv;
  cv.v = __builtin_amdgcn_cvt_pk_bf16_f32(lo, hi);
  return cv.u;
#else
  return ((unsigned)f2bf(hi) << 16) | f2bf(lo);
#endif
}

// DPP16 row-rotate move: pure-VALU cross-lane within a 16-lane row (no LDS).
template <int CTRL>
__device__ __forceinline__ float dpp_mov(float x) {
  return __int_as_float(
      __builtin_amdgcn_update_dpp(0, __float_as_int(x), CTRL, 0xf, 0xf, true));
}
// Rotate-based all-reduce over the 16-lane half: row_ror 1,2,4,8 (0x121..0x128).
__device__ __forceinline__ float half_max16(float x) {
  x = fmaxf(x, dpp_mov<0x121>(x));
  x = fmaxf(x, dpp_mov<0x122>(x));
  x = fmaxf(x, dpp_mov<0x124>(x));
  x = fmaxf(x, dpp_mov<0x128>(x));
  return x;
}
__device__ __forceinline__ float half_sum16(float x) {
  x += dpp_mov<0x121>(x);
  x += dpp_mov<0x122>(x);
  x += dpp_mov<0x124>(x);
  x += dpp_mov<0x128>(x);
  return x;
}

// ---------------------------------------------------------------------------
// Kernel 1: tiled bf16 WMMA GEMM (e @ c^T) with online logsumexp per row.
// grid = (N/BM, SLICES); block = 256 (8 waves, wave32).
// Register double-buffered global->LDS staging; DPP epilogue reductions.
// ---------------------------------------------------------------------------
__launch_bounds__(256)
__global__ void cce_lse_kernel(const float* __restrict__ E,
                               const float* __restrict__ C,
                               float* __restrict__ pM,
                               float* __restrict__ pS) {
  __shared__ unsigned short shA[BM * LDA];
  __shared__ unsigned short shB[BN * LDA];
  __shared__ float sh_m[BM], sh_s[BM], sh_newm[BM];
  __shared__ float sh_tmax[BM][2], sh_tsum[BM][2];

  const int tid   = threadIdx.x;
  const int lane  = tid & 31;
  const int wid   = tid >> 5;
  const int waveM = wid & 3;    // 4 row bands of 32
  const int waveN = wid >> 2;   // 2 col bands of 64
  const int half  = lane >> 4;  // 0: lanes 0-15, 1: lanes 16-31
  const int lr    = lane & 15;

  const int rowBase   = blockIdx.x * BM;
  const int sliceBase = blockIdx.y * VSLICE;

  // Per-thread staging coordinates (8 float4 quads per matrix per stage).
  const int sr  = tid >> 4;         // base row for it==0 (rows advance by 16/it)
  const int scq = tid & 15;         // quad column within the 64-wide K chunk

  if (tid < BM) { sh_m[tid] = -INFINITY; sh_s[tid] = 0.f; }
  __syncthreads();

  for (int tile = 0; tile < VSLICE / BN; ++tile) {
    const int colBase = sliceBase + tile * BN;

    v8f acc[2][4];
#pragma unroll
    for (int mi = 0; mi < 2; ++mi)
#pragma unroll
      for (int ni = 0; ni < 4; ++ni)
#pragma unroll
        for (int k = 0; k < 8; ++k) acc[mi][ni][k] = 0.f;

    // ---- software-pipelined K loop: regs(s+1) load overlaps compute(s) ----
    float4 regA[8], regB[8];
#pragma unroll
    for (int it = 0; it < 8; ++it) {           // prologue: stage 0 -> regs
      const int r = sr + it * 16;
      regA[it] = *(const float4*)(E + (size_t)(rowBase + r) * Ddim + scq * 4);
      regB[it] = *(const float4*)(C + (size_t)(colBase + r) * Ddim + scq * 4);
    }

    for (int s = 0; s < KSTAGES; ++s) {
      __syncthreads();                         // LDS consumers of stage s-1 done
#pragma unroll
      for (int it = 0; it < 8; ++it) {         // convert + store stage s to LDS
        const int r = sr + it * 16;
        uint2 pa, pb;
        pa.x = pack_bf16(regA[it].x, regA[it].y);
        pa.y = pack_bf16(regA[it].z, regA[it].w);
        pb.x = pack_bf16(regB[it].x, regB[it].y);
        pb.y = pack_bf16(regB[it].z, regB[it].w);
        *(uint2*)&shA[r * LDA + scq * 4] = pa;
        *(uint2*)&shB[r * LDA + scq * 4] = pb;
      }
      __syncthreads();

      if (s + 1 < KSTAGES) {                   // prefetch stage s+1 into regs
        const int k0 = (s + 1) * KC;
#pragma unroll
        for (int it = 0; it < 8; ++it) {
          const int r = sr + it * 16;
          regA[it] = *(const float4*)(E + (size_t)(rowBase + r) * Ddim + k0 + scq * 4);
          regB[it] = *(const float4*)(C + (size_t)(colBase + r) * Ddim + k0 + scq * 4);
        }
      }

#pragma unroll
      for (int ks = 0; ks < KC; ks += 32) {
        // Fragment loads per ISA 16-bit A/B layout:
        // half 0 holds K = ks+0..7 and ks+16..23; half 1 holds K = ks+8..15 and ks+24..31.
        FragBF a[2], b[4];
#pragma unroll
        for (int mi = 0; mi < 2; ++mi) {
          const unsigned short* p = &shA[(waveM * 32 + mi * 16 + lr) * LDA + ks + half * 8];
          a[mi].u[0] = *(const uint4*)p;       // 16B-aligned ds_load_b128
          a[mi].u[1] = *(const uint4*)(p + 16);
        }
#pragma unroll
        for (int ni = 0; ni < 4; ++ni) {
          const unsigned short* p = &shB[(waveN * 64 + ni * 16 + lr) * LDA + ks + half * 8];
          b[ni].u[0] = *(const uint4*)p;
          b[ni].u[1] = *(const uint4*)(p + 16);
        }
#pragma unroll
        for (int mi = 0; mi < 2; ++mi)
#pragma unroll
          for (int ni = 0; ni < 4; ++ni)
            acc[mi][ni] = __builtin_amdgcn_wmma_f32_16x16x32_bf16(
                false, a[mi].v, false, b[ni].v, (short)0, acc[mi][ni], false, false);
      }
    }

    // ---- online logsumexp fold of this 128x128 logit tile ----
    // C/D layout: VGPR v, lanes 0-15 -> M = base+v; lanes 16-31 -> M = base+v+8.
#pragma unroll
    for (int mi = 0; mi < 2; ++mi) {
#pragma unroll
      for (int v = 0; v < 8; ++v) {
        float x = fmaxf(fmaxf(acc[mi][0][v], acc[mi][1][v]),
                        fmaxf(acc[mi][2][v], acc[mi][3][v]));
        x = half_max16(x);                     // DPP, no LDS
        if (lr == 0) sh_tmax[waveM * 32 + mi * 16 + v + 8 * half][waveN] = x;
      }
    }
    __syncthreads();
    if (tid < BM)
      sh_newm[tid] = fmaxf(sh_m[tid], fmaxf(sh_tmax[tid][0], sh_tmax[tid][1]));
    __syncthreads();
#pragma unroll
    for (int mi = 0; mi < 2; ++mi) {
#pragma unroll
      for (int v = 0; v < 8; ++v) {
        const int row = waveM * 32 + mi * 16 + v + 8 * half;
        const float nm = sh_newm[row];
        float x = __expf(acc[mi][0][v] - nm) + __expf(acc[mi][1][v] - nm)
                + __expf(acc[mi][2][v] - nm) + __expf(acc[mi][3][v] - nm);
        x = half_sum16(x);                     // DPP, no LDS
        if (lr == 0) sh_tsum[row][waveN] = x;
      }
    }
    __syncthreads();
    if (tid < BM) {
      const float nm = sh_newm[tid];
      sh_s[tid] = sh_s[tid] * __expf(sh_m[tid] - nm) + sh_tsum[tid][0] + sh_tsum[tid][1];
      sh_m[tid] = nm;
    }
    __syncthreads();
  }

  if (tid < BM) {
    pM[(size_t)blockIdx.y * Ntok + rowBase + tid] = sh_m[tid];
    pS[(size_t)blockIdx.y * Ntok + rowBase + tid] = sh_s[tid];
  }
}

// ---------------------------------------------------------------------------
// Kernel 2: target logits t[n] = dot(e[n], c[targets[n]]) in fp32 (cheap).
// ---------------------------------------------------------------------------
__launch_bounds__(256)
__global__ void cce_tgt_kernel(const float* __restrict__ E,
                               const float* __restrict__ C,
                               const int* __restrict__ T,
                               float* __restrict__ tgt) {
  __shared__ float red[8];
  const int row = blockIdx.x;
  const int tid = threadIdx.x;
  const int t = T[row];
  float p = 0.f;
  if (t != IGNORE_IDX) {
    const float* er = E + (size_t)row * Ddim;
    const float* cr = C + (size_t)t * Ddim;
    for (int k = tid * 4; k < Ddim; k += 256 * 4) {
      float4 a = *(const float4*)(er + k);
      float4 b = *(const float4*)(cr + k);
      p += a.x * b.x + a.y * b.y + a.z * b.z + a.w * b.w;
    }
  }
#pragma unroll
  for (int m = 16; m >= 1; m >>= 1) p += __shfl_xor(p, m, 32);
  if ((tid & 31) == 0) red[tid >> 5] = p;
  __syncthreads();
  if (tid == 0) {
    float s = 0.f;
#pragma unroll
    for (int i = 0; i < 8; ++i) s += red[i];
    tgt[row] = s;
  }
}

// ---------------------------------------------------------------------------
// Kernel 3: combine slice partials -> lse, masked mean NLL -> out[0].
// ---------------------------------------------------------------------------
__launch_bounds__(256)
__global__ void cce_final_kernel(const float* __restrict__ pM,
                                 const float* __restrict__ pS,
                                 const float* __restrict__ tgt,
                                 const int* __restrict__ T,
                                 float* __restrict__ out) {
  __shared__ float rsum[256];
  __shared__ int   rcnt[256];
  const int tid = threadIdx.x;
  float lsum = 0.f;
  int lcnt = 0;
  for (int row = tid; row < Ntok; row += 256) {
    if (T[row] == IGNORE_IDX) continue;
    float m = -INFINITY;
    for (int s = 0; s < SLICES; ++s) m = fmaxf(m, pM[(size_t)s * Ntok + row]);
    float ss = 0.f;
    for (int s = 0; s < SLICES; ++s)
      ss += pS[(size_t)s * Ntok + row] * __expf(pM[(size_t)s * Ntok + row] - m);
    lsum += (m + __logf(ss)) - tgt[row];
    lcnt += 1;
  }
  rsum[tid] = lsum;
  rcnt[tid] = lcnt;
  __syncthreads();
  for (int off = 128; off >= 1; off >>= 1) {
    if (tid < off) { rsum[tid] += rsum[tid + off]; rcnt[tid] += rcnt[tid + off]; }
    __syncthreads();
  }
  if (tid == 0) {
    const int c = rcnt[0] < 1 ? 1 : rcnt[0];
    out[0] = rsum[0] / (float)c;
  }
}

// ---------------------------------------------------------------------------
extern "C" void kernel_launch(void* const* d_in, const int* in_sizes, int n_in,
                              void* d_out, int out_size, void* d_ws, size_t ws_size,
                              hipStream_t stream) {
  const float* E = (const float*)d_in[0];   // [N, D] fp32
  const float* C = (const float*)d_in[1];   // [V, D] fp32
  const int*   T = (const int*)d_in[2];     // [N] int
  float* out = (float*)d_out;               // scalar fp32

  // workspace: pM [SLICES*N] | pS [SLICES*N] | tgt [N]  (~1.06 MB)
  float* pM  = (float*)d_ws;
  float* pS  = pM + (size_t)SLICES * Ntok;
  float* tgl = pS + (size_t)SLICES * Ntok;

  cce_lse_kernel<<<dim3(Ntok / BM, SLICES), 256, 0, stream>>>(E, C, pM, pS);
  cce_tgt_kernel<<<Ntok, 256, 0, stream>>>(E, C, T, tgl);
  cce_final_kernel<<<1, 256, 0, stream>>>(pM, pS, tgl, T, out);
}
